// LSTM_12378095747446
// MI455X (gfx1250) — matrix operand order repaired and checked by reading, not compiled, not run on hardware
//
#include <hip/hip_runtime.h>

// ---------------- problem constants ----------------
#define TT   512
#define BSZ  64
#define ISZ  1024
#define HSZ  1024
#define NHT  (HSZ / 16)   // 64 column tiles per gate
#define NKT  (HSZ / 32)   // 32 K-tiles (K-step 32 for bf16 WMMA)

typedef __bf16          v16bf __attribute__((ext_vector_type(16)));
typedef float           v8f   __attribute__((ext_vector_type(8)));
typedef unsigned short  v16us __attribute__((ext_vector_type(16)));
typedef unsigned int    v4u   __attribute__((ext_vector_type(4)));
typedef unsigned short  u16;

union FragB {
    v16bf bf;
    v16us us;
    v4u   q[2];
};

__device__ inline u16 f2bf(float x) {
    unsigned int u = __float_as_uint(x);
    return (u16)((u + 0x7FFFu + ((u >> 16) & 1u)) >> 16);
}

__device__ inline float sigmoidf_(float x) {
    return 1.0f / (1.0f + __expf(-x));
}

__device__ inline v8f wmma_bf16(v16bf a, v16bf b, v8f c) {
    return __builtin_amdgcn_wmma_f32_16x16x32_bf16(
        /*neg_a=*/false, a, /*neg_b=*/false, b,
        /*c_mod=*/(short)0, c, /*reuse_a=*/false, /*reuse_b=*/false);
}

// A-matrix 16x32 bf16 fragment (ISA 7.12.2): lanes 0-15 -> rows M=0..15 with
// K {0..7, 16..23}; lanes 16-31 -> same rows with K {8..15, 24..31}.
__device__ inline v16bf load_frag_a(const u16* base, int ld,
                                    int r0, int k0, int lane) {
    int l = lane & 15, hi = lane >> 4;
    const u16* p = base + (size_t)(r0 + l) * ld + k0 + hi * 8;
    FragB f;
    f.q[0] = *(const v4u*)p;
    f.q[1] = *(const v4u*)(p + 16);
    return f.bf;
}

// B-matrix 32x16 bf16 fragment: lanes 0-15 -> cols N=0..15 with K {0..7, 16..23},
// lanes 16-31 -> same cols with K {8..15, 24..31} (matches A fragment).
// Weight stored row-major [4H, K]; B[k][n] = W[n][k].
__device__ inline v16bf load_frag_b(const u16* __restrict__ base, int ld,
                                    int n0, int k0, int lane) {
    int l = lane & 15, hi = lane >> 4;
    const u16* p = base + (size_t)(n0 + l) * ld + k0 + hi * 8;
    FragB f;
    f.q[0] = *(const v4u*)p;
    f.q[1] = *(const v4u*)(p + 16);
    return f.bf;
}

// ---------------- software-pipelined 4-gate GEMM pass ----------------
// Accumulates x_g += A(16xK) x W_g(Kx16) for the four gate column blocks,
// with explicit register double-buffering: iteration kt issues loads for
// kt+1 before consuming kt's fragments, so WMMAs only wait on loads that
// are one full iteration old (loads return in order -> partial loadcnt waits).
__device__ inline void gemm4_pipelined(const u16* Ab, const u16* __restrict__ Wb,
                                       int ld, int r0, int nh, int lane,
                                       v8f& x0, v8f& x1, v8f& x2, v8f& x3) {
    const int n0 = nh * 16;
    v16bf a  = load_frag_a(Ab, ld, r0, 0, lane);
    v16bf b0 = load_frag_b(Wb, ld, 0 * HSZ + n0, 0, lane);
    v16bf b1 = load_frag_b(Wb, ld, 1 * HSZ + n0, 0, lane);
    v16bf b2 = load_frag_b(Wb, ld, 2 * HSZ + n0, 0, lane);
    v16bf b3 = load_frag_b(Wb, ld, 3 * HSZ + n0, 0, lane);
#pragma unroll 4
    for (int kt = 0; kt < NKT; ++kt) {
        const int k1 = ((kt + 1) & (NKT - 1)) * 32;   // last iter: dead reload
        __builtin_prefetch(Wb + (size_t)n0 * ld + k1 + 256, 0, 1);
        v16bf an = load_frag_a(Ab, ld, r0, k1, lane);
        v16bf n0f = load_frag_b(Wb, ld, 0 * HSZ + n0, k1, lane);
        v16bf n1f = load_frag_b(Wb, ld, 1 * HSZ + n0, k1, lane);
        v16bf n2f = load_frag_b(Wb, ld, 2 * HSZ + n0, k1, lane);
        v16bf n3f = load_frag_b(Wb, ld, 3 * HSZ + n0, k1, lane);
        x0 = wmma_bf16(a, b0, x0);
        x1 = wmma_bf16(a, b1, x1);
        x2 = wmma_bf16(a, b2, x2);
        x3 = wmma_bf16(a, b3, x3);
        a = an; b0 = n0f; b1 = n1f; b2 = n2f; b3 = n3f;
    }
}

// ---------------- helper kernels ----------------
__global__ void cvt_bf16_kernel(const float* __restrict__ src,
                                u16* __restrict__ dst, int n) {
    int i = blockIdx.x * blockDim.x + threadIdx.x;
    int stride = gridDim.x * blockDim.x;
    for (; i < n; i += stride) dst[i] = f2bf(src[i]);
}

__global__ void zero_kernel(unsigned int* __restrict__ p, int nwords) {
    int i = blockIdx.x * blockDim.x + threadIdx.x;
    int stride = gridDim.x * blockDim.x;
    for (; i < nwords; i += stride) p[i] = 0u;
}

// ---------------- time-parallel GEMM: X0 = x @ Wx0^T ----------------
// wave -> (mg = M-tile over T*B, nh = h-column tile); one A fragment feeds
// 4 WMMAs (all gates). Output in WMMA C-fragment layout:
//   tile index = ((t*4 + gate)*NHT + nh)*4 + mt, 256 floats per tile.
__global__ __launch_bounds__(256) void xgemm_kernel(
    const u16* __restrict__ xb, const u16* __restrict__ Wx0b,
    float* __restrict__ x0frag) {
    const int lane = threadIdx.x & 31;
    const int wave = (int)((blockIdx.x * 256u + threadIdx.x) >> 5);
    const int mg = wave >> 6;                 // 0..2047  (M-tile over T*B)
    const int nh = wave & 63;                 // 0..63    (h-column tile)

    v8f a0 = {0.f,0.f,0.f,0.f,0.f,0.f,0.f,0.f};
    v8f a1 = a0, a2 = a0, a3 = a0;
    gemm4_pipelined(xb, Wx0b, ISZ, mg * 16, nh, lane, a0, a1, a2, a3);

    const int t = mg >> 2, mt = mg & 3;
    size_t base = (((size_t)t * 4) * NHT + nh) * 4 + mt;
    *(v8f*)(x0frag + (base + 0 * NHT * 4) * 256 + lane * 8) = a0;
    *(v8f*)(x0frag + (base + 1 * NHT * 4) * 256 + lane * 8) = a1;
    *(v8f*)(x0frag + (base + 2 * NHT * 4) * 256 + lane * 8) = a2;
    *(v8f*)(x0frag + (base + 3 * NHT * 4) * 256 + lane * 8) = a3;
}

// ---------------- device-wide barrier (persistent kernel) ----------------
__device__ inline void grid_sync(int* bar) {
    __threadfence();
    __syncthreads();
    if (threadIdx.x == 0) {
        int* cnt = bar;
        int* gen = bar + 16;
        int g0 = __hip_atomic_load(gen, __ATOMIC_RELAXED, __HIP_MEMORY_SCOPE_AGENT);
        int arrived = __hip_atomic_fetch_add(cnt, 1, __ATOMIC_ACQ_REL,
                                             __HIP_MEMORY_SCOPE_AGENT) + 1;
        if (arrived == (int)gridDim.x) {
            __hip_atomic_store(cnt, 0, __ATOMIC_RELAXED, __HIP_MEMORY_SCOPE_AGENT);
            __hip_atomic_fetch_add(gen, 1, __ATOMIC_RELEASE, __HIP_MEMORY_SCOPE_AGENT);
        } else {
            while (__hip_atomic_load(gen, __ATOMIC_ACQUIRE,
                                     __HIP_MEMORY_SCOPE_AGENT) == g0)
                __builtin_amdgcn_s_sleep(1);
        }
    }
    __syncthreads();
    __threadfence();
}

// ---------------- persistent recurrent kernel: all 512 steps ----------------
// 64 blocks x 128 threads = 256 waves, 4 waves/block -> 1 wave per SIMD32 so
// the serial WMMA chain never shares a matrix pipe. wave -> (nh, mt); each
// wave owns a 16x16 tile for ALL FOUR gates, cell update fused in registers.
// Cell state c is kept in C-fragment layout ((wave*32+lane)*8 floats) since it
// is wave-private: one 32B vector load + store per layer instead of 16 strided
// dwords. h0 double-buffered -> single device barrier per step.
__global__ __launch_bounds__(128, 1) void lstm_seq_kernel(
    const float* __restrict__ x0frag,
    const u16* __restrict__ Wh0b, const u16* __restrict__ Wx1b,
    const u16* __restrict__ Wh1b,
    u16* h0buf0, u16* h0buf1, u16* h1b,
    float* __restrict__ c0f, float* __restrict__ c1f,
    float* __restrict__ out, int* bar) {
    const int lane = threadIdx.x & 31;
    const int wave = (int)((blockIdx.x * 128u + threadIdx.x) >> 5);  // 0..255
    const int nh = wave >> 2;     // 0..63
    const int mt = wave & 3;      // 0..3
    const int l = lane & 15, hi = lane >> 4;
    const int ncol = nh * 16 + l;
    const size_t coff = ((size_t)wave * 32 + lane) * 8;  // private c fragment
    const size_t THB = (size_t)TT * BSZ * HSZ;
    const size_t BH = (size_t)BSZ * HSZ;

    for (int t = 0; t < TT; ++t) {
        u16* h0w = (t & 1) ? h0buf1 : h0buf0;        // A(t) writes here
        const u16* h0r = (t & 1) ? h0buf0 : h0buf1;  // holds h0(t-1)

        // ================= layer 0: ifgo0 = X0[t] + h0 @ Wh0^T =============
        v8f acc0, acc1, acc2, acc3;
        {
            size_t base = (((size_t)t * 4) * NHT + nh) * 4 + mt;
            acc0 = *(const v8f*)(x0frag + (base + 0 * NHT * 4) * 256 + lane * 8);
            acc1 = *(const v8f*)(x0frag + (base + 1 * NHT * 4) * 256 + lane * 8);
            acc2 = *(const v8f*)(x0frag + (base + 2 * NHT * 4) * 256 + lane * 8);
            acc3 = *(const v8f*)(x0frag + (base + 3 * NHT * 4) * 256 + lane * 8);
        }
        gemm4_pipelined(h0r, Wh0b, HSZ, mt * 16, nh, lane,
                        acc0, acc1, acc2, acc3);
        {
            v8f cp = *(const v8f*)(c0f + coff);
            v8f cnv;
#pragma unroll
            for (int r = 0; r < 8; ++r) {
                int m = mt * 16 + hi * 8 + r;   // C layout: lanes>=16 hold M+8
                size_t idx = (size_t)m * HSZ + ncol;
                float ig = sigmoidf_(acc0[r]);
                float jg = tanhf(acc1[r]);
                float fg = sigmoidf_(acc2[r] + 1.0f);   // FORGET_BIAS
                float og = sigmoidf_(acc3[r]);
                float cn = cp[r] * fg + ig * jg;
                cnv[r] = cn;
                float hn = og * tanhf(cn);
                h0w[idx] = f2bf(hn);
                if (t == TT - 1) {
                    out[THB + idx] = hn;            // h[0]
                    out[THB + 2 * BH + idx] = cn;   // c[0]
                }
            }
            *(v8f*)(c0f + coff) = cnv;
        }
        grid_sync(bar);

        // ============ layer 1: ifgo1 = h0 @ Wx1^T + h1 @ Wh1^T =============
        // Two sequential pipelined passes into the same accumulators keeps
        // live fragment count low (no spills) while preserving the pipeline.
        v8f b0 = {0.f,0.f,0.f,0.f,0.f,0.f,0.f,0.f};
        v8f b1 = b0, b2 = b0, b3 = b0;
        gemm4_pipelined(h0w, Wx1b, HSZ, mt * 16, nh, lane, b0, b1, b2, b3);
        gemm4_pipelined(h1b, Wh1b, HSZ, mt * 16, nh, lane, b0, b1, b2, b3);
        {
            v8f cp = *(const v8f*)(c1f + coff);
            v8f cnv;
#pragma unroll
            for (int r = 0; r < 8; ++r) {
                int m = mt * 16 + hi * 8 + r;
                size_t idx = (size_t)m * HSZ + ncol;
                float ig = sigmoidf_(b0[r]);
                float jg = tanhf(b1[r]);
                float fg = sigmoidf_(b2[r] + 1.0f);
                float og = sigmoidf_(b3[r]);
                float cn = cp[r] * fg + ig * jg;
                cnv[r] = cn;
                float hn = og * tanhf(cn);
                h1b[idx] = f2bf(hn);
                out[(size_t)(t * BSZ + m) * HSZ + ncol] = hn;   // out[t]
                if (t == TT - 1) {
                    out[THB + BH + idx] = hn;           // h[1]
                    out[THB + 3 * BH + idx] = cn;       // c[1]
                }
            }
            *(v8f*)(c1f + coff) = cnv;
        }
        // no barrier here: next step's A->B barrier orders B(t) vs B(t+1)
    }
}

// ---------------- host entry ----------------
extern "C" void kernel_launch(void* const* d_in, const int* in_sizes, int n_in,
                              void* d_out, int out_size, void* d_ws, size_t ws_size,
                              hipStream_t stream) {
    (void)in_sizes; (void)n_in; (void)out_size; (void)ws_size;
    const float* x   = (const float*)d_in[0];
    const float* Wx0 = (const float*)d_in[1];
    const float* Wh0 = (const float*)d_in[2];
    const float* Wx1 = (const float*)d_in[3];
    const float* Wh1 = (const float*)d_in[4];
    float* out = (float*)d_out;

    const size_t WELT = (size_t)4 * HSZ * ISZ;   // 4,194,304 elems per weight
    const size_t XELT = (size_t)TT * BSZ * ISZ;  // 33,554,432 elems of x
    const size_t BH   = (size_t)BSZ * HSZ;

    char* ws = (char*)d_ws;
    u16* Wx0b = (u16*)ws;
    u16* Wh0b = Wx0b + WELT;
    u16* Wx1b = Wh0b + WELT;
    u16* Wh1b = Wx1b + WELT;
    u16* xb   = Wh1b + WELT;                      // x in bf16, row-major
    float* c0f = (float*)(xb + XELT);             // c in fragment layout
    float* c1f = c0f + BH;
    u16* h0b0 = (u16*)(c1f + BH);
    u16* h0b1 = h0b0 + BH;
    u16* h1b  = h0b1 + BH;
    int* bar  = (int*)(h1b + BH);                 // 128B barrier slot
    float* x0frag = (float*)((char*)bar + 128);   // T*B*4H fp32, fragment layout

    // 1) weights + x -> bf16 (weights stay resident in 192MB L2)
    cvt_bf16_kernel<<<2048, 256, 0, stream>>>(Wx0, Wx0b, (int)WELT);
    cvt_bf16_kernel<<<2048, 256, 0, stream>>>(Wh0, Wh0b, (int)WELT);
    cvt_bf16_kernel<<<2048, 256, 0, stream>>>(Wx1, Wx1b, (int)WELT);
    cvt_bf16_kernel<<<2048, 256, 0, stream>>>(Wh1, Wh1b, (int)WELT);
    cvt_bf16_kernel<<<8192, 256, 0, stream>>>(x, xb, (int)XELT);

    // 2) zero recurrent state + barrier
    size_t state_bytes = (size_t)((char*)x0frag - (char*)c0f);
    zero_kernel<<<256, 256, 0, stream>>>((unsigned int*)c0f,
                                         (int)(state_bytes / 4));

    // 3) time-parallel GEMM: X0 = x @ Wx0^T
    xgemm_kernel<<<16384, 256, 0, stream>>>(xb, Wx0b, x0frag);

    // 4) persistent recurrence: all 512 steps, one launch
    //    64 blocks x 4 waves -> one wave per SIMD32 across 64 WGPs
    lstm_seq_kernel<<<64, 128, 0, stream>>>(x0frag, Wh0b, Wx1b, Wh1b,
                                            h0b0, h0b1, h1b, c0f, c1f, out, bar);
}
